// KalmanNet_GRU_12532714569982
// MI455X (gfx1250) — matrix operand order repaired and checked by reading, not compiled, not run on hardware
//
#include <hip/hip_runtime.h>

// ---------------------------------------------------------------------------
// KalmanNet GRU on MI455X (gfx1250): per-step WMMA f16 GEMMs, f32 state carry.
// B=2048, Q=64, D=256, H=512, w_out=32
// Round 5: hardware V_TANH_F32 for tanh/sigmoid epilogues (guarded), on top of
//          the GLOBAL_LOAD_ASYNC_TO_LDS double-buffered WMMA pipeline.
// ---------------------------------------------------------------------------

typedef _Float16 half_t;
typedef __attribute__((ext_vector_type(16))) _Float16 v16h;
typedef __attribute__((ext_vector_type(8)))  _Float16 v8h;
typedef __attribute__((ext_vector_type(8)))  float    v8f;
typedef __attribute__((ext_vector_type(4)))  int      v4i;

typedef __attribute__((address_space(1))) v4i as1_v4i;
typedef __attribute__((address_space(3))) v4i as3_v4i;

#if __has_builtin(__builtin_amdgcn_global_load_async_to_lds_b128)
#define USE_ASYNC_LDS 1
#else
#define USE_ASYNC_LDS 0
#endif

__device__ __forceinline__ void wait_async0() {
#if __has_builtin(__builtin_amdgcn_s_wait_asynccnt)
    __builtin_amdgcn_s_wait_asynccnt(0);
#else
    asm volatile("s_wait_asynccnt 0x0" ::: "memory");
#endif
}

// hardware tanh (V_TANH_F32) when available; sigmoid via exact tanh identity
__device__ __forceinline__ float fast_tanh(float x) {
#if __has_builtin(__builtin_amdgcn_tanhf)
    return __builtin_amdgcn_tanhf(x);
#else
    return tanhf(x);
#endif
}

__device__ __forceinline__ float sigf(float x) {
#if __has_builtin(__builtin_amdgcn_tanhf)
    return 0.5f * __builtin_amdgcn_tanhf(0.5f * x) + 0.5f;
#else
    return 1.0f / (1.0f + __expf(-x));
#endif
}

#define B_   2048
#define Q_   64
#define D_   256
#define H_   512
#define WOUT 32

// ---------------------------------------------------------------------------
// WMMA GEMM: C = act(A(MxK,f16) * W(NxK,f16)^T + bias [+ addend])
// 128x128 block tile, 8 waves (wave32), each wave 32x64 (2x4 of 16x16 frags),
// BK=32, double-buffered LDS fed by GLOBAL_LOAD_ASYNC_TO_LDS_B128.
// ACT: 0 = none, 1 = tanh, 2 = sigmoid.  Output config is compile-time.
// ---------------------------------------------------------------------------
template <int ACT, bool HAS_ADD, bool OUT32, bool OUT16, bool OUT2>
__global__ __launch_bounds__(256)
void gemm_wmma(const half_t* __restrict__ A, const half_t* __restrict__ W,
               const float* __restrict__ bias,
               const float* addend, int ld_add,
               float* outF32, int ldo32,
               half_t* outF16, int ldo16,
               float* out2, int ldo2,
               int M, int N, int K)
{
    // two LDS buffers; rows padded to 48 halves (96B, 16B-aligned)
    __shared__ half_t As[2][128 * 48];
    __shared__ half_t Bs[2][128 * 48];

    const int tid  = threadIdx.x;
    const int lane = tid & 31;
    const int wave = tid >> 5;
    const int wm   = wave & 3;        // wave row -> M offset wm*32
    const int wn   = wave >> 2;       // wave col -> N offset wn*64
    const int bM   = blockIdx.x * 128;
    const int bN   = blockIdx.y * 128;

    v8f acc[2][4];
#pragma unroll
    for (int i = 0; i < 2; ++i)
#pragma unroll
        for (int j = 0; j < 4; ++j)
            acc[i][j] = (v8f){0.f, 0.f, 0.f, 0.f, 0.f, 0.f, 0.f, 0.f};

    // cooperative-load geometry: 512 chunks of 8 halves per tile, 2 per thread
    int rowA[2], ccA[2], ldsOff[2];
    const half_t* Aptr[2];
    const half_t* Wptr[2];
#pragma unroll
    for (int i = 0; i < 2; ++i) {
        const int c = tid * 2 + i;            // 0..511
        rowA[i]   = c >> 2;                   // 0..127
        ccA[i]    = (c & 3) * 8;              // 0,8,16,24
        ldsOff[i] = rowA[i] * 48 + ccA[i];
        Aptr[i]   = A + (size_t)(bM + rowA[i]) * K + ccA[i];
        Wptr[i]   = W + (size_t)(bN + rowA[i]) * K + ccA[i];
    }

    // fragment addressing (ISA 16-bit A 16x32 / B 32x16 lane layout)
    const int am = lane & 15;
    const int ak = (lane >> 4) * 8;
    const int bn = lane & 15;
    const int bk = (lane >> 4) * 8;

    const int NK = K >> 5;                    // K-chunks of 32 (K % 32 == 0)

#if USE_ASYNC_LDS
    // prologue: async-copy chunk 0 straight into LDS buffer 0
#pragma unroll
    for (int i = 0; i < 2; ++i) {
        __builtin_amdgcn_global_load_async_to_lds_b128(
            (as1_v4i*)Aptr[i], (as3_v4i*)&As[0][ldsOff[i]], 0, 0);
        __builtin_amdgcn_global_load_async_to_lds_b128(
            (as1_v4i*)Wptr[i], (as3_v4i*)&Bs[0][ldsOff[i]], 0, 0);
    }
    wait_async0();
    __syncthreads();
#else
    v8h sa[2], sb[2];
#pragma unroll
    for (int i = 0; i < 2; ++i) {
        sa[i] = *(const v8h*)(Aptr[i]);
        sb[i] = *(const v8h*)(Wptr[i]);
    }
#pragma unroll
    for (int i = 0; i < 2; ++i) {
        *(v8h*)&As[0][ldsOff[i]] = sa[i];
        *(v8h*)&Bs[0][ldsOff[i]] = sb[i];
    }
    __syncthreads();
#endif

    for (int kc = 0; kc < NK; ++kc) {
        const int cur = kc & 1;

        // prefetch the K-stream two chunks ahead (global_prefetch_b8)
        if (kc + 2 < NK) {
            __builtin_prefetch(Aptr[0] + (size_t)(kc + 2) * 32, 0, 1);
            __builtin_prefetch(Wptr[0] + (size_t)(kc + 2) * 32, 0, 1);
        }

        // overlap: move chunk kc+1 into the other buffer while computing kc
#if USE_ASYNC_LDS
        if (kc + 1 < NK) {
#pragma unroll
            for (int i = 0; i < 2; ++i) {
                __builtin_amdgcn_global_load_async_to_lds_b128(
                    (as1_v4i*)(Aptr[i] + (size_t)(kc + 1) * 32),
                    (as3_v4i*)&As[cur ^ 1][ldsOff[i]], 0, 0);
                __builtin_amdgcn_global_load_async_to_lds_b128(
                    (as1_v4i*)(Wptr[i] + (size_t)(kc + 1) * 32),
                    (as3_v4i*)&Bs[cur ^ 1][ldsOff[i]], 0, 0);
            }
        }
#else
        if (kc + 1 < NK) {
#pragma unroll
            for (int i = 0; i < 2; ++i) {
                sa[i] = *(const v8h*)(Aptr[i] + (size_t)(kc + 1) * 32);
                sb[i] = *(const v8h*)(Wptr[i] + (size_t)(kc + 1) * 32);
            }
        }
#endif

        // fragments from LDS buffer `cur`
        v16h afr[2], bfr[4];
#pragma unroll
        for (int mt = 0; mt < 2; ++mt) {
            const half_t* p = &As[cur][(wm * 32 + mt * 16 + am) * 48 + ak];
            v8h lo = *(const v8h*)p;
            v8h hi = *(const v8h*)(p + 16);
            afr[mt] = __builtin_shufflevector(lo, hi,
                0,1,2,3,4,5,6,7,8,9,10,11,12,13,14,15);
        }
#pragma unroll
        for (int nt = 0; nt < 4; ++nt) {
            const half_t* p = &Bs[cur][(wn * 64 + nt * 16 + bn) * 48 + bk];
            v8h lo = *(const v8h*)p;
            v8h hi = *(const v8h*)(p + 16);
            bfr[nt] = __builtin_shufflevector(lo, hi,
                0,1,2,3,4,5,6,7,8,9,10,11,12,13,14,15);
        }
#pragma unroll
        for (int mt = 0; mt < 2; ++mt)
#pragma unroll
            for (int nt = 0; nt < 4; ++nt)
                acc[mt][nt] = __builtin_amdgcn_wmma_f32_16x16x32_f16(
                    false, afr[mt], false, bfr[nt], (short)0, acc[mt][nt],
                    false, false);

        // hand the freshly filled buffer to the next iteration
        if (kc + 1 < NK) {
#if USE_ASYNC_LDS
            wait_async0();
#else
#pragma unroll
            for (int i = 0; i < 2; ++i) {
                *(v8h*)&As[cur ^ 1][ldsOff[i]] = sa[i];
                *(v8h*)&Bs[cur ^ 1][ldsOff[i]] = sb[i];
            }
#endif
            __syncthreads();
        }
    }

    // branch-free epilogue (all config compile-time)
    const int r0 = (lane >> 4) * 8;
    const int cn = lane & 15;
#pragma unroll
    for (int mt = 0; mt < 2; ++mt) {
#pragma unroll
        for (int nt = 0; nt < 4; ++nt) {
            const int gcol = bN + wn * 64 + nt * 16 + cn;
            const float bv = bias[gcol];
            const size_t rbase = (size_t)(bM + wm * 32 + mt * 16 + r0);
#pragma unroll
            for (int v = 0; v < 8; ++v) {
                const size_t grow = rbase + v;
                float x = acc[mt][nt][v] + bv;
                if constexpr (HAS_ADD) x += addend[grow * ld_add + gcol];
                if constexpr (ACT == 1) x = fast_tanh(x);
                if constexpr (ACT == 2) x = sigf(x);
                if constexpr (OUT32) outF32[grow * ldo32 + gcol] = x;
                if constexpr (OUT16) outF16[grow * ldo16 + gcol] = (half_t)x;
                if constexpr (OUT2)  out2[grow * ldo2 + gcol]    = x;
            }
        }
    }
}

// ---------------------------------------------------------------------------
// Elementwise / small kernels
// ---------------------------------------------------------------------------
__global__ void k_f32_to_f16(const float* __restrict__ in, half_t* __restrict__ out, int n) {
    int i = blockIdx.x * blockDim.x + threadIdx.x;
    if (i < n) out[i] = (half_t)in[i];
}

__global__ void k_zero_f32(float* p, int n) {
    int i = blockIdx.x * blockDim.x + threadIdx.x;
    if (i < n) p[i] = 0.0f;
}

__global__ void k_zero_f16(half_t* p, int n) {
    int i = blockIdx.x * blockDim.x + threadIdx.x;
    if (i < n) p[i] = (half_t)0.0f;
}

__global__ void k_copy_cast(const float* __restrict__ in, float* __restrict__ o32,
                            half_t* __restrict__ o16, int n) {
    int i = blockIdx.x * blockDim.x + threadIdx.x;
    if (i < n) { float v = in[i]; o32[i] = v; o16[i] = (half_t)v; }
}

// feat = concat([y_t - x_post, x_post, y_t - y_prev]) -> f16 (B x 768)
__global__ void k_build_feat(const float* __restrict__ xin, const float* __restrict__ xp,
                             half_t* __restrict__ feat, int t) {
    int idx = blockIdx.x * blockDim.x + threadIdx.x;
    if (idx >= B_ * 3 * D_) return;
    int b = idx / (3 * D_), k = idx % (3 * D_);
    const float* y  = xin + (size_t)b * Q_ * D_ + (size_t)t * D_;
    const int tp = (t > 0) ? (t - 1) : 0;    // scan carry init: y_prev = xs[0]
    const float* yp = xin + (size_t)b * Q_ * D_ + (size_t)tp * D_;
    float v;
    if (k < D_)           v = y[k] - xp[(size_t)b * D_ + k];
    else if (k < 2 * D_)  v = xp[(size_t)b * D_ + (k - D_)];
    else                  v = y[k - 2 * D_] - yp[k - 2 * D_];
    feat[idx] = (half_t)v;
}

// fused GRU gate update: h = (1-z)*n + z*h  (gi, gh are B x 3H)
__global__ void k_gru_update(const float* __restrict__ gi, const float* __restrict__ gh,
                             float* __restrict__ h, half_t* __restrict__ h16) {
    int idx = blockIdx.x * blockDim.x + threadIdx.x;
    if (idx >= B_ * H_) return;
    int b = idx / H_, j = idx % H_;
    size_t o = (size_t)b * 3 * H_;
    float ir = gi[o + j], iz = gi[o + H_ + j], in = gi[o + 2 * H_ + j];
    float hr = gh[o + j], hz = gh[o + H_ + j], hn = gh[o + 2 * H_ + j];
    float r = sigf(ir + hr);
    float z = sigf(iz + hz);
    float n = fast_tanh(in + r * hn);
    float hv = (1.0f - z) * n + z * h[idx];
    h[idx]   = hv;
    h16[idx] = (half_t)hv;
}

// rank-2 GEMV: rp = h @ fc_rot_w^T + b; cs[b] = (rho*cos(phi), rho*sin(phi))
__global__ __launch_bounds__(256)
void k_rot_gemv(const float* __restrict__ h, const float* __restrict__ w,
                const float* __restrict__ bias, float* __restrict__ cs) {
    int gwave = (blockIdx.x * 256 + threadIdx.x) >> 5;
    int lane  = threadIdx.x & 31;
    if (gwave >= B_) return;
    const float* hr = h + (size_t)gwave * H_;
    float s0 = 0.0f, s1 = 0.0f;
    for (int k = lane; k < H_; k += 32) {
        float hv = hr[k];
        s0 += hv * w[k];
        s1 += hv * w[H_ + k];
    }
#pragma unroll
    for (int off = 16; off > 0; off >>= 1) {
        s0 += __shfl_xor(s0, off, 32);
        s1 += __shfl_xor(s1, off, 32);
    }
    if (lane == 0) {
        float rho = 1.5f * sigf(s0 + bias[0]);
        float phi = fast_tanh(s1 + bias[1]) * 3.14159265358979323846f;
        cs[gwave * 2 + 0] = rho * __cosf(phi);
        cs[gwave * 2 + 1] = rho * __sinf(phi);
    }
}

// Kalman update: x_pri = rotate(x_post); x_post = x_pri + K*(y - x_pri)
// one thread owns the (re, im) pair -> no in-place race.
__global__ void k_update_x(const float* __restrict__ xin, const float* __restrict__ Kg,
                           const float* __restrict__ cs, float* __restrict__ xp, int t) {
    int idx = blockIdx.x * blockDim.x + threadIdx.x;
    const int M = D_ / 2;
    if (idx >= B_ * M) return;
    int b = idx / M, m = idx % M;
    const float* y = xin + (size_t)b * Q_ * D_ + (size_t)t * D_;
    float c = cs[b * 2 + 0], s = cs[b * 2 + 1];
    float xr = xp[(size_t)b * D_ + m];
    float xi = xp[(size_t)b * D_ + M + m];
    float pr = xr * c - xi * s;
    float pi = xr * s + xi * c;
    float k0 = Kg[(size_t)b * D_ + m];
    float k1 = Kg[(size_t)b * D_ + M + m];
    xp[(size_t)b * D_ + m]     = pr + k0 * (y[m]     - pr);
    xp[(size_t)b * D_ + M + m] = pi + k1 * (y[M + m] - pi);
}

// ---------------------------------------------------------------------------
// Host-side orchestration
// ---------------------------------------------------------------------------
extern "C" void kernel_launch(void* const* d_in, const int* in_sizes, int n_in,
                              void* d_out, int out_size, void* d_ws, size_t ws_size,
                              hipStream_t stream) {
    const float* x_in      = (const float*)d_in[0];
    const float* fc_feat_w = (const float*)d_in[1];
    const float* fc_feat_b = (const float*)d_in[2];
    const float* gru_wih   = (const float*)d_in[3];
    const float* gru_whh   = (const float*)d_in[4];
    const float* gru_bih   = (const float*)d_in[5];
    const float* gru_bhh   = (const float*)d_in[6];
    const float* fc_gain_w = (const float*)d_in[7];
    const float* fc_gain_b = (const float*)d_in[8];
    const float* fc_rot_w  = (const float*)d_in[9];
    const float* fc_rot_b  = (const float*)d_in[10];
    const float* grup_wih  = (const float*)d_in[11];
    const float* grup_whh  = (const float*)d_in[12];
    const float* grup_bih  = (const float*)d_in[13];
    const float* grup_bhh  = (const float*)d_in[14];
    const float* fc_out_w  = (const float*)d_in[15];
    const float* fc_out_b  = (const float*)d_in[16];
    float* out = (float*)d_out;

    // ---- workspace bump allocator (256B aligned) ----
    char* base = (char*)d_ws;
    size_t pos = 0;
    auto alloc = [&](size_t bytes) -> void* {
        void* p = base + pos;
        pos = (pos + bytes + 255) & ~(size_t)255;
        return p;
    };

    // f16 weight shadows
    half_t* wfeat16 = (half_t*)alloc((size_t)H_ * 3 * D_ * 2);
    half_t* wih16   = (half_t*)alloc((size_t)3 * H_ * H_ * 2);
    half_t* whh16   = (half_t*)alloc((size_t)3 * H_ * H_ * 2);
    half_t* wgain16 = (half_t*)alloc((size_t)D_ * H_ * 2);
    half_t* pwih16  = (half_t*)alloc((size_t)3 * H_ * D_ * 2);
    half_t* pwhh16  = (half_t*)alloc((size_t)3 * H_ * H_ * 2);
    half_t* wout16  = (half_t*)alloc((size_t)D_ * H_ * 2);
    // activations / state
    half_t* feat16 = (half_t*)alloc((size_t)B_ * 3 * D_ * 2);
    half_t* z16    = (half_t*)alloc((size_t)B_ * H_ * 2);
    float*  h32    = (float*) alloc((size_t)B_ * H_ * 4);
    half_t* h16    = (half_t*)alloc((size_t)B_ * H_ * 2);
    float*  gi32   = (float*) alloc((size_t)B_ * 3 * H_ * 4);
    float*  gh32   = (float*) alloc((size_t)B_ * 3 * H_ * 4);
    float*  xp32   = (float*) alloc((size_t)B_ * D_ * 4);
    float*  Kg32   = (float*) alloc((size_t)B_ * D_ * 4);
    float*  cs     = (float*) alloc((size_t)B_ * 2 * 4);
    float*  hp32   = (float*) alloc((size_t)B_ * H_ * 4);
    half_t* hp16   = (half_t*)alloc((size_t)B_ * H_ * 2);
    float*  curr32 = (float*) alloc((size_t)B_ * D_ * 4);
    half_t* curr16 = (half_t*)alloc((size_t)B_ * D_ * 2);
    (void)ws_size; (void)in_sizes; (void)n_in; (void)out_size;

    const int T = 256;
    auto blks = [](int n) { return (n + 255) / 256; };

    // ---- one-time (per launch) weight conversion + state init ----
    k_f32_to_f16<<<blks(H_ * 3 * D_), T, 0, stream>>>(fc_feat_w, wfeat16, H_ * 3 * D_);
    k_f32_to_f16<<<blks(3 * H_ * H_), T, 0, stream>>>(gru_wih,   wih16,   3 * H_ * H_);
    k_f32_to_f16<<<blks(3 * H_ * H_), T, 0, stream>>>(gru_whh,   whh16,   3 * H_ * H_);
    k_f32_to_f16<<<blks(D_ * H_),     T, 0, stream>>>(fc_gain_w, wgain16, D_ * H_);
    k_f32_to_f16<<<blks(3 * H_ * D_), T, 0, stream>>>(grup_wih,  pwih16,  3 * H_ * D_);
    k_f32_to_f16<<<blks(3 * H_ * H_), T, 0, stream>>>(grup_whh,  pwhh16,  3 * H_ * H_);
    k_f32_to_f16<<<blks(D_ * H_),     T, 0, stream>>>(fc_out_w,  wout16,  D_ * H_);
    k_zero_f32<<<blks(B_ * H_), T, 0, stream>>>(h32, B_ * H_);
    k_zero_f16<<<blks(B_ * H_), T, 0, stream>>>(h16, B_ * H_);
    k_zero_f32<<<blks(B_ * D_), T, 0, stream>>>(xp32, B_ * D_);

    // ---- filter loop: 64 sequential steps ----
    for (int t = 0; t < Q_; ++t) {
        k_build_feat<<<blks(B_ * 3 * D_), T, 0, stream>>>(x_in, xp32, feat16, t);

        // z = tanh(feat @ Wfeat^T + b) -> f16       M=2048 N=512 K=768
        gemm_wmma<1, false, false, true, false>
            <<<dim3(B_ / 128, H_ / 128), T, 0, stream>>>(
            feat16, wfeat16, fc_feat_b, nullptr, 0,
            nullptr, 0, z16, H_, nullptr, 0, B_, H_, 3 * D_);

        // gi = z @ Wih^T + bih                      M=2048 N=1536 K=512
        gemm_wmma<0, false, true, false, false>
            <<<dim3(B_ / 128, (3 * H_) / 128), T, 0, stream>>>(
            z16, wih16, gru_bih, nullptr, 0,
            gi32, 3 * H_, nullptr, 0, nullptr, 0, B_, 3 * H_, H_);

        // gh = h @ Whh^T + bhh                      M=2048 N=1536 K=512
        gemm_wmma<0, false, true, false, false>
            <<<dim3(B_ / 128, (3 * H_) / 128), T, 0, stream>>>(
            h16, whh16, gru_bhh, nullptr, 0,
            gh32, 3 * H_, nullptr, 0, nullptr, 0, B_, 3 * H_, H_);

        k_gru_update<<<blks(B_ * H_), T, 0, stream>>>(gi32, gh32, h32, h16);

        k_rot_gemv<<<B_ / 8, T, 0, stream>>>(h32, fc_rot_w, fc_rot_b, cs);

        // K = sigmoid(h @ Wgain^T + b)              M=2048 N=256 K=512
        gemm_wmma<2, false, true, false, false>
            <<<dim3(B_ / 128, D_ / 128), T, 0, stream>>>(
            h16, wgain16, fc_gain_b, nullptr, 0,
            Kg32, D_, nullptr, 0, nullptr, 0, B_, D_, H_);

        k_update_x<<<blks(B_ * (D_ / 2)), T, 0, stream>>>(x_in, Kg32, cs, xp32, t);
    }

    // ---- prediction loop: 32 sequential steps ----
    k_copy_cast<<<blks(B_ * D_), T, 0, stream>>>(xp32, curr32, curr16, B_ * D_);
    k_zero_f32<<<blks(B_ * H_), T, 0, stream>>>(hp32, B_ * H_);
    k_zero_f16<<<blks(B_ * H_), T, 0, stream>>>(hp16, B_ * H_);

    for (int t = 0; t < WOUT; ++t) {
        // gi = curr @ pWih^T + bih                  M=2048 N=1536 K=256
        gemm_wmma<0, false, true, false, false>
            <<<dim3(B_ / 128, (3 * H_) / 128), T, 0, stream>>>(
            curr16, pwih16, grup_bih, nullptr, 0,
            gi32, 3 * H_, nullptr, 0, nullptr, 0, B_, 3 * H_, D_);

        // gh = hp @ pWhh^T + bhh                    M=2048 N=1536 K=512
        gemm_wmma<0, false, true, false, false>
            <<<dim3(B_ / 128, (3 * H_) / 128), T, 0, stream>>>(
            hp16, pwhh16, grup_bhh, nullptr, 0,
            gh32, 3 * H_, nullptr, 0, nullptr, 0, B_, 3 * H_, H_);

        k_gru_update<<<blks(B_ * H_), T, 0, stream>>>(gi32, gh32, hp32, hp16);

        // curr = curr + hp @ Wout^T + b; also write preds[:, t, :]
        gemm_wmma<0, true, true, true, true>
            <<<dim3(B_ / 128, D_ / 128), T, 0, stream>>>(
            hp16, wout16, fc_out_b, curr32, D_,
            curr32, D_, curr16, D_, out + (size_t)t * D_, WOUT * D_,
            B_, D_, H_);
    }
}
